// ResidualVQ_64905545777869
// MI455X (gfx1250) — compile-verified, matrix operand order checked by reading, not run on hardware
//
#include <hip/hip_runtime.h>
#include <hip/hip_bf16.h>
#include <stdint.h>

// Problem constants (match reference)
#define B_    32
#define N_    2048
#define D_    512
#define Q_    8
#define C_    1024
#define ROWS  (B_ * N_)        // 65536
#define COMMIT_W 0.02f

// Tiling
#define THREADS 256            // 8 wave32s
#define WAVES   8
#define M_TILE  128            // rows per block (16 per wave)
#define KCHUNKS (D_ / 32)      // 16
#define NTILES  (C_ / 16)      // 64
#define NBLOCKS (ROWS / M_TILE) // 512
#define BPREF   (16 * D_ / THREADS)   // 32 codebook elements prefetched per thread

typedef __attribute__((ext_vector_type(16))) __bf16 v16bf;
typedef __attribute__((ext_vector_type(8)))  float  v8f;

__device__ __forceinline__ unsigned short f32_to_bf16_trunc(float f) {
  union { float f; unsigned u; } c; c.f = f;
  return (unsigned short)(c.u >> 16);
}
__device__ __forceinline__ float bf16us_to_f32(unsigned short h) {
  union { float f; unsigned u; } c; c.u = ((unsigned)h) << 16;
  return c.f;
}

// Map a K-offset within a 32-wide chunk to (lane-half, element) of the
// 16-bit A/B fragment layout (ISA 7.12.2): lanes 0-15 hold K 0..7,16..23,
// lanes 16-31 hold K 8..15,24..31; element e packs K-pairs per VGPR.
__device__ __forceinline__ void k_to_frag(int ko, int& hi16, int& e) {
  hi16 = (ko >> 3) & 1;
  e    = (ko & 7) | ((ko & 16) >> 1);
}

// LDS element counts (ushorts)
#define A_US   (M_TILE * D_)
#define B_US   (16 * D_)

__global__ void rvq_stage_kernel(const float* __restrict__ codebook,   // this stage's (C_, D_)
                                 float* __restrict__ residual,         // (ROWS, D_) in/out
                                 float* __restrict__ out_idx,          // (ROWS, Q_) as float
                                 float* __restrict__ loss_partials,    // (NBLOCKS,)
                                 int qi)
{
  extern __shared__ __align__(16) char smem[];
  unsigned short* Ahi = (unsigned short*)smem;
  unsigned short* Alo = Ahi + A_US;
  unsigned short* Bhi = Alo + A_US;
  unsigned short* Blo = Bhi + B_US;
  float* enorm   = (float*)(Blo + B_US);       // 16 floats
  int*   bestidx = (int*)(enorm + 16);         // M_TILE ints
  float* red     = (float*)(bestidx + M_TILE); // THREADS floats

  const int tid  = threadIdx.x;
  const int lane = tid & 31;
  const int wave = tid >> 5;
  const int row0 = blockIdx.x * M_TILE;

  // ---------- Stage A: residual rows -> split bf16 hi/lo in fragment layout ----------
  for (int j = tid; j < M_TILE * D_; j += THREADS) {
    int m = j >> 9;                // row within tile (D_ = 512)
    int k = j & (D_ - 1);
    float r = residual[(size_t)(row0 + m) * D_ + k];
    unsigned short hs = f32_to_bf16_trunc(r);
    unsigned short ls = f32_to_bf16_trunc(r - bf16us_to_f32(hs));
    int kc = k >> 5, ko = k & 31, hi16, e;
    k_to_frag(ko, hi16, e);
    int lds_lane = hi16 * 16 + (m & 15);
    int off = (m >> 4) * (16 * 512) + kc * 512 + lds_lane * 16 + e;
    Ahi[off] = hs;
    Alo[off] = ls;
  }

  // Per-thread B staging geometry (same element set every tile)
  int pn[BPREF], poff[BPREF];
#pragma unroll
  for (int i = 0; i < BPREF; ++i) {
    int j = tid + i * THREADS;
    int n = j >> 9;
    int k = j & (D_ - 1);
    int kc = k >> 5, ko = k & 31, hi16, e;
    k_to_frag(ko, hi16, e);
    pn[i]   = n;
    poff[i] = kc * 512 + (hi16 * 16 + n) * 16 + e;
  }

  // Prefetch B tile 0 into registers
  float pref[BPREF];
#pragma unroll
  for (int i = 0; i < BPREF; ++i) {
    int j = tid + i * THREADS;
    pref[i] = codebook[(size_t)(j >> 9) * D_ + (j & (D_ - 1))];
  }

  float minv[8];
  int   mini[8];
#pragma unroll
  for (int v = 0; v < 8; ++v) { minv[v] = 3.4e38f; mini[v] = 0; }

  // ---------- Sweep codebook in tiles of 16 codes ----------
  for (int nt = 0; nt < NTILES; ++nt) {
    const int n0 = nt * 16;
    __syncthreads();                       // previous tile fully consumed
    if (tid < 16) enorm[tid] = 0.f;
    __syncthreads();

    // Stage B tile from prefetch registers: hi/lo bf16 fragments + ||e||^2
#pragma unroll
    for (int i = 0; i < BPREF; ++i) {
      float v = pref[i];
      unsigned short hs = f32_to_bf16_trunc(v);
      unsigned short ls = f32_to_bf16_trunc(v - bf16us_to_f32(hs));
      Bhi[poff[i]] = hs;
      Blo[poff[i]] = ls;
      atomicAdd(&enorm[pn[i]], v * v);
    }
    __syncthreads();

    // Prefetch next tile's codebook values; loads retire under the WMMA burst
    if (nt + 1 < NTILES) {
#pragma unroll
      for (int i = 0; i < BPREF; ++i) {
        int j = tid + i * THREADS;
        pref[i] = codebook[(size_t)((nt + 1) * 16 + (j >> 9)) * D_ + (j & (D_ - 1))];
      }
    }

    // 16x16 tile of dot products, K = 512, split-bf16 3-product scheme.
    // Three independent accumulators -> adjacent WMMAs have no RAW chain.
    v8f acc0 = {0.f, 0.f, 0.f, 0.f, 0.f, 0.f, 0.f, 0.f};
    v8f acc1 = {0.f, 0.f, 0.f, 0.f, 0.f, 0.f, 0.f, 0.f};
    v8f acc2 = {0.f, 0.f, 0.f, 0.f, 0.f, 0.f, 0.f, 0.f};
#pragma unroll
    for (int kc = 0; kc < KCHUNKS; ++kc) {
      const int abase = wave * (16 * 512) + kc * 512 + lane * 16;
      const int bbase = kc * 512 + lane * 16;
      v16bf a_hi = *(const v16bf*)(Ahi + abase);
      v16bf a_lo = *(const v16bf*)(Alo + abase);
      v16bf b_hi = *(const v16bf*)(Bhi + bbase);
      v16bf b_lo = *(const v16bf*)(Blo + bbase);
      acc0 = __builtin_amdgcn_wmma_f32_16x16x32_bf16(false, a_hi, false, b_hi,
                                                     (short)0, acc0, false, false);
      acc1 = __builtin_amdgcn_wmma_f32_16x16x32_bf16(false, a_hi, false, b_lo,
                                                     (short)0, acc1, false, false);
      acc2 = __builtin_amdgcn_wmma_f32_16x16x32_bf16(false, a_lo, false, b_hi,
                                                     (short)0, acc2, false, false);
    }

    // score = ||e||^2 - 2 r.e   (||r||^2 constant per row -> irrelevant to argmin)
    float en   = enorm[lane & 15];
    int   nidx = n0 + (lane & 15);
#pragma unroll
    for (int v = 0; v < 8; ++v) {
      float dot = acc0[v] + acc1[v] + acc2[v];
      float score = en - 2.f * dot;
      if (score < minv[v]) { minv[v] = score; mini[v] = nidx; }
    }
  }

  // ---------- Cross-lane argmin within each 16-lane half (C/D layout: N = lane&15) ----------
#pragma unroll
  for (int v = 0; v < 8; ++v) {
#pragma unroll
    for (int off = 8; off >= 1; off >>= 1) {
      float ov = __shfl_xor(minv[v], off, 16);
      int   oi = __shfl_xor(mini[v], off, 16);
      if (ov < minv[v] || (ov == minv[v] && oi < mini[v])) { minv[v] = ov; mini[v] = oi; }
    }
  }
  if ((lane & 15) == 0) {
    int half = lane >> 4;                 // M = v + 8*half within this wave's strip
#pragma unroll
    for (int v = 0; v < 8; ++v) bestidx[wave * 16 + 8 * half + v] = mini[v];
  }
  __syncthreads();

  // ---------- Epilogue: indices, residual update, commit-loss partial ----------
  if (tid < M_TILE) {
    out_idx[(size_t)(row0 + tid) * Q_ + qi] = (float)bestidx[tid];
  }

  float lsum = 0.f;
  for (int j = tid; j < M_TILE * D_; j += THREADS) {
    int m = j >> 9;
    int k = j & (D_ - 1);
    int ci = bestidx[m];
    float ev = codebook[(size_t)ci * D_ + k];
    size_t gi = (size_t)(row0 + m) * D_ + k;
    float rn = residual[gi] - ev;        // new residual; (q - r)^2 == rn^2
    residual[gi] = rn;
    lsum += rn * rn;
  }
  red[tid] = lsum;
  __syncthreads();
  for (int s = THREADS / 2; s > 0; s >>= 1) {
    if (tid < s) red[tid] += red[tid + s];
    __syncthreads();
  }
  if (tid == 0) loss_partials[blockIdx.x] = red[0];
}

// Deterministic single-block reduction of per-block loss partials.
__global__ void rvq_loss_reduce_kernel(const float* __restrict__ partials,
                                       float* __restrict__ loss_all,
                                       float* __restrict__ loss_final,
                                       int qi)
{
  __shared__ float red[256];
  float s = 0.f;
  for (int i = threadIdx.x; i < NBLOCKS; i += 256) s += partials[i];
  red[threadIdx.x] = s;
  __syncthreads();
  for (int st = 128; st > 0; st >>= 1) {
    if (threadIdx.x < st) red[threadIdx.x] += red[threadIdx.x + st];
    __syncthreads();
  }
  if (threadIdx.x == 0) {
    float v = red[0] * (COMMIT_W / ((float)ROWS * (float)D_));
    loss_all[qi] = v;
    loss_final[0] += v;     // sequential single-block kernels -> deterministic
  }
}

__global__ void rvq_init_kernel(const float* __restrict__ x,
                                float* __restrict__ residual,
                                float* __restrict__ losses /* 9 floats */)
{
  size_t i = (size_t)blockIdx.x * blockDim.x + threadIdx.x;
  size_t stride = (size_t)gridDim.x * blockDim.x;
  for (size_t j = i; j < (size_t)ROWS * D_; j += stride) residual[j] = x[j];
  if (i < 9) losses[i] = 0.f;
}

__global__ void rvq_final_kernel(const float* __restrict__ x,
                                 float* __restrict__ outq /* holds final residual */)
{
  size_t i = (size_t)blockIdx.x * blockDim.x + threadIdx.x;
  size_t stride = (size_t)gridDim.x * blockDim.x;
  for (size_t j = i; j < (size_t)ROWS * D_; j += stride) outq[j] = x[j] - outq[j];
}

extern "C" void kernel_launch(void* const* d_in, const int* in_sizes, int n_in,
                              void* d_out, int out_size, void* d_ws, size_t ws_size,
                              hipStream_t stream) {
  const float* x         = (const float*)d_in[0];   // (B,N,D) f32
  const float* codebooks = (const float*)d_in[1];   // (Q,C,D) f32

  float* out        = (float*)d_out;
  float* residual   = out;                               // quantized_out region doubles as residual
  float* out_idx    = out + (size_t)ROWS * D_;           // (ROWS, Q)
  float* loss_all   = out_idx + (size_t)ROWS * Q_;       // (Q,)
  float* loss_final = loss_all + Q_;                     // scalar
  float* ws         = (float*)d_ws;                      // NBLOCKS floats of loss partials

  const size_t smem_bytes =
      (size_t)A_US * 2 * sizeof(unsigned short) +        // Ahi + Alo
      (size_t)B_US * 2 * sizeof(unsigned short) +        // Bhi + Blo
      16 * sizeof(float) + M_TILE * sizeof(int) + THREADS * sizeof(float);

  rvq_init_kernel<<<1024, 256, 0, stream>>>(x, residual, loss_all);

  for (int qi = 0; qi < Q_; ++qi) {
    const float* cb = codebooks + (size_t)qi * C_ * D_;
    rvq_stage_kernel<<<NBLOCKS, THREADS, smem_bytes, stream>>>(
        cb, residual, out_idx, ws, qi);
    rvq_loss_reduce_kernel<<<1, 256, 0, stream>>>(ws, loss_all, loss_final, qi);
  }

  rvq_final_kernel<<<1024, 256, 0, stream>>>(x, out);
}
